// AttentionMV_5179730559504
// MI455X (gfx1250) — matrix operand order repaired and checked by reading, not compiled, not run on hardware
//
#include <hip/hip_runtime.h>
#include <hip/hip_bf16.h>

typedef __bf16 bf16_t;
typedef bf16_t v16bf __attribute__((ext_vector_type(16)));
typedef float  v8f   __attribute__((ext_vector_type(8)));

#define T_DIM   1024
#define E_DIM   128
#define CHUNK   32            // j rows per iteration (= WMMA K of GEMM2)
#define NCHUNK  (T_DIM / CHUNK)
#define MSTR    136           // LDS row stride (bf16) for m chunk  (272B = 17*16B, b128-aligned)
#define WSTR    136           // LDS row stride (bf16) for W^T
#define VSTR    40            // LDS row stride (bf16) for v^T chunk (80B = 5*16B)
#define PSTR    132           // LDS row stride (f32) for pre tiles

// LDS union layout (bytes)
#define W_OFF   0
#define W_BYTES (E_DIM * WSTR * 2)        // 34816
#define M_OFF   (W_OFF + W_BYTES)
#define M_BYTES (CHUNK * MSTR * 2)        // 8704
#define V_OFF   (M_OFF + M_BYTES)
#define V_BYTES (E_DIM * VSTR * 2)        // 10240
#define UNION_BYTES (V_OFF + V_BYTES)     // 53760 ; tail reuses [0, 64*PSTR*4)=33792 for pre

static __device__ __forceinline__ float fast_tanh(float x) {
#if __has_builtin(__builtin_amdgcn_tanhf)
  return __builtin_amdgcn_tanhf(x);        // gfx1250 V_TANH_F32
#else
  float e = __expf(2.0f * x);
  return 1.0f - 2.0f / (e + 1.0f);
#endif
}

static __device__ __forceinline__ unsigned pack2(float a, float b) {
  union { bf16_t h[2]; unsigned u; } x;
  x.h[0] = (bf16_t)a; x.h[1] = (bf16_t)b;
  return x.u;
}

static __device__ __forceinline__ v16bf ldfrag(const bf16_t* p0, const bf16_t* p1) {
  v16bf r;
  uint4* u = reinterpret_cast<uint4*>(&r);
  u[0] = *reinterpret_cast<const uint4*>(p0);   // elements 0..7
  u[1] = *reinterpret_cast<const uint4*>(p1);   // elements 8..15
  return r;
}

__global__ __launch_bounds__(256)
void attn_mv_fused(const float* __restrict__ m, const float* __restrict__ v,
                   const float* __restrict__ W, const float* __restrict__ b,
                   float* __restrict__ out) {
  __shared__ alignas(16) unsigned char smem[UNION_BYTES];
  __shared__ float red256[256];
  __shared__ float msumLds[E_DIM];
  __shared__ float rowMax[64];
  __shared__ float rowScale[64];

  bf16_t* wlds   = (bf16_t*)(smem + W_OFF);   // W^T : [f=128][e=128] bf16, stride WSTR
  bf16_t* mlds   = (bf16_t*)(smem + M_OFF);   // m chunk : [j=32][e=128] bf16, stride MSTR
  bf16_t* vlds   = (bf16_t*)(smem + V_OFF);   // v^T chunk : [e=128][j=32] bf16, stride VSTR
  float*  preLds = (float*)smem;              // tail: pre half [64][128] f32, stride PSTR

  const int tid  = threadIdx.x;
  const int wave = tid >> 5;          // 0..7, owns output cols [16*wave, 16*wave+16)
  const int lane = tid & 31;
  const int hlf  = lane >> 4;         // lane half (wave32)
  const int l16  = lane & 15;
  const int bb   = blockIdx.x;
  const int fcol = 16 * wave + l16;

  const float* mB = m + (size_t)bb * T_DIM * E_DIM;
  const float* WB = W + (size_t)bb * E_DIM * E_DIM;

  // ---- stage W_b transposed as bf16 into LDS (one time) ----
  #pragma unroll 4
  for (int k = 0; k < 64; ++k) {
    int idx = tid + 256 * k;                  // 0..16383, coalesced global read
    int e = idx >> 7, f = idx & 127;
    wlds[f * WSTR + e] = (bf16_t)WB[idx];
  }
  __syncthreads();

  // ---- preload this wave's W B-fragments: B[K=e(32s..32s+31), N=fcol strip] ----
  // B layout: element p of lane L holds K = 32s + 16*hlf + p, N = l16  -> 16 contiguous bf16
  v16bf wfrag[4];
  {
    const bf16_t* wrow = wlds + (16 * wave + l16) * WSTR + 16 * hlf;
    #pragma unroll
    for (int s = 0; s < 4; ++s)
      wfrag[s] = ldfrag(wrow + 32 * s, wrow + 32 * s + 8);
  }

  // pre accumulators: 8 e-tiles x (16x16 f32) for this wave's column strip
  v8f preAcc[8];
  #pragma unroll
  for (int r = 0; r < 8; ++r) preAcc[r] = v8f{0,0,0,0,0,0,0,0};

  float msumAcc = 0.0f;   // per-thread partial of sum_t m[b,t,e], e == tid&127

  for (int c = 0; c < NCHUNK; ++c) {
    const int j0 = c * CHUNK;
    __syncthreads();
    if (c + 1 < NCHUNK && tid < 128)  // prefetch next m chunk (128 cachelines)
      __builtin_prefetch(mB + (size_t)(j0 + CHUNK) * E_DIM + tid * 32, 0, 1);

    // ---- stage m chunk (bf16, row-major) and v^T chunk (bf16) ----
    #pragma unroll 4
    for (int k = 0; k < 16; ++k) {
      int idx = tid + 256 * k;        // 0..4095; e = idx&127 == tid&127 (constant per thread)
      int jl = idx >> 7, e = idx & 127;
      float mv = mB[(size_t)(j0 + jl) * E_DIM + e];
      msumAcc += mv;
      mlds[jl * MSTR + e] = (bf16_t)mv;
      vlds[e * VSTR + jl] = (bf16_t)v[(size_t)(j0 + jl) * E_DIM + e];
    }
    __syncthreads();

    // ---- GEMM1: u(32 x 16strip) = m_chunk(32x128) @ W(128 x 16strip) ----
    // A layout: element p of lane L -> row M=l16(+16t), K = 32s + 16*hlf + p
    v8f u0 = v8f{0,0,0,0,0,0,0,0};
    v8f u1 = v8f{0,0,0,0,0,0,0,0};
    #pragma unroll
    for (int s = 0; s < 4; ++s) {
      const bf16_t* a0 = mlds + l16 * MSTR        + 32 * s + 16 * hlf;
      const bf16_t* a1 = mlds + (16 + l16) * MSTR + 32 * s + 16 * hlf;
      v16bf af0 = ldfrag(a0, a0 + 8);
      v16bf af1 = ldfrag(a1, a1 + 8);
      u0 = __builtin_amdgcn_wmma_f32_16x16x32_bf16(false, af0, false, wfrag[s],
                                                   (short)0, u0, false, false);
      u1 = __builtin_amdgcn_wmma_f32_16x16x32_bf16(false, af1, false, wfrag[s],
                                                   (short)0, u1, false, false);
    }

    // ---- +b, tanh, repack D-tiles into GEMM2's B fragment (half-lane swap) ----
    // D: u0 vgpr g -> row j0+g+8*hlf ; u1 vgpr g -> row j0+16+g+8*hlf ; col fcol
    float t0[8], t1[8];
    #pragma unroll
    for (int g = 0; g < 8; ++g) {
      int r0 = j0 + g + 8 * hlf;
      t0[g] = fast_tanh(u0[g] + b[(size_t)r0 * E_DIM + fcol]);
      t1[g] = fast_tanh(u1[g] + b[(size_t)(r0 + 16) * E_DIM + fcol]);
    }
    unsigned pk0[4], pk1[4], q0[4], q1[4];
    #pragma unroll
    for (int i = 0; i < 4; ++i) {
      pk0[i] = pack2(t0[2*i], t0[2*i+1]);
      pk1[i] = pack2(t1[2*i], t1[2*i+1]);
      q0[i] = (unsigned)__shfl_xor((int)pk0[i], 16, 32);
      q1[i] = (unsigned)__shfl_xor((int)pk1[i], 16, 32);
    }
    // B needs rows K = 16*hlf + p at col l16
    v16bf bu;
    unsigned* bp = reinterpret_cast<unsigned*>(&bu);
    #pragma unroll
    for (int i = 0; i < 4; ++i) {
      bp[i]     = hlf ? q1[i]  : pk0[i];   // p=0..7
      bp[i + 4] = hlf ? pk1[i] : q0[i];    // p=8..15
    }

    // ---- GEMM2: pre(128 x 16strip) += v^T chunk(128x32) @ u chunk(32 x 16strip) ----
    #pragma unroll
    for (int r = 0; r < 8; ++r) {
      const bf16_t* ap = vlds + (16 * r + l16) * VSTR + 16 * hlf;
      v16bf af = ldfrag(ap, ap + 16 - 8);
      preAcc[r] = __builtin_amdgcn_wmma_f32_16x16x32_bf16(false, af, false, bu,
                                                          (short)0, preAcc[r], false, false);
    }
  }

  // ---- reduce msum: threads tid and tid+128 both own e = tid&127 ----
  __syncthreads();
  red256[tid] = msumAcc;
  __syncthreads();
  if (tid < 128) msumLds[tid] = red256[tid] + red256[tid + 128];

  // ---- softmax rows + final matvec, in two 64-row halves of pre ----
  float outV = 0.0f;
  #pragma unroll
  for (int hs = 0; hs < 2; ++hs) {
    __syncthreads();
    #pragma unroll
    for (int g2 = 0; g2 < 4; ++g2) {
      int r = 4 * hs + g2;
      #pragma unroll
      for (int g = 0; g < 8; ++g) {
        int el = 16 * g2 + g + 8 * hlf;               // e - 64*hs
        preLds[el * PSTR + fcol] = preAcc[r][g];
      }
    }
    __syncthreads();
    if (tid < 64) {                                   // per-row max & sum-exp
      const float* row = preLds + tid * PSTR;
      float mx = -3.4e38f;
      for (int j = 0; j < 128; ++j) mx = fmaxf(mx, row[j]);
      float s = 0.0f;
      for (int j = 0; j < 128; ++j) s += __expf(row[j] - mx);
      rowMax[tid] = mx;
      rowScale[tid] = msumLds[64 * hs + tid] / s;
    }
    __syncthreads();
    if (tid < 128) {                                  // out[m] += sum_e scale_e*exp(...)
      float acc = 0.0f;
      for (int e = 0; e < 64; ++e)
        acc += rowScale[e] * __expf(preLds[e * PSTR + tid] - rowMax[e]);
      outV += acc;
    }
  }
  if (tid < 128) out[(size_t)bb * E_DIM + tid] = outV;
}

extern "C" void kernel_launch(void* const* d_in, const int* in_sizes, int n_in,
                              void* d_out, int out_size, void* d_ws, size_t ws_size,
                              hipStream_t stream) {
  const float* m = (const float*)d_in[0];   // (B,T,E) f32
  const float* v = (const float*)d_in[1];   // (B,E)   f32
  const float* W = (const float*)d_in[2];   // (B,E,E) f32
  const float* b = (const float*)d_in[3];   // (B,E)   f32
  float* out = (float*)d_out;               // (B,E)   f32
  (void)in_sizes; (void)n_in; (void)out_size; (void)d_ws; (void)ws_size;

  attn_mv_fused<<<dim3(1024), dim3(256), 0, stream>>>(m, v, W, b, out);
}
